// SelfAttnBlock_19061064859982
// MI455X (gfx1250) — matrix operand order repaired and checked by reading, not compiled, not run on hardware
//
#include <hip/hip_runtime.h>

// Problem constants (from reference): B=4, C=256, W=H=64 -> N=4096, CQK=32
#define BB   4
#define CC   256
#define NN   4096
#define CQK  32

typedef unsigned short u16;
typedef u16    v8u   __attribute__((ext_vector_type(8)));
typedef u16    v16u  __attribute__((ext_vector_type(16)));
typedef __bf16 v16bf __attribute__((ext_vector_type(16)));
typedef float  v8f   __attribute__((ext_vector_type(8)));
typedef float  v16f  __attribute__((ext_vector_type(16)));
typedef unsigned int v4u32 __attribute__((ext_vector_type(4)));
typedef int    v8i32 __attribute__((ext_vector_type(8)));
typedef int    v4i32 __attribute__((ext_vector_type(4)));

// float -> bf16 bits, round-to-nearest-even
__device__ __forceinline__ u16 f2bf(float f) {
  unsigned u = __builtin_bit_cast(unsigned, f);
  unsigned r = u + 0x7FFFu + ((u >> 16) & 1u);
  return (u16)(r >> 16);
}

// Build WMMA A-operand (16x32 bf16) for one lane from a bf16 row pointer.
// Layout: lane<16 holds K {0..15}, lane>=16 holds K {16..31} (matches B operand).
__device__ __forceinline__ v16bf load_a16(const u16* __restrict__ row, int half) {
  const v16u au = *(const v16u*)(row + half * 16);
  return __builtin_bit_cast(v16bf, au);
}

// ---------------------------------------------------------------------------
// K1: Q/K projection. out[b][n][o] = sum_c w[o][c]*x[b][c][n] + bias[o], bf16.
// Output layout (B*N rows x 32) is directly the WMMA A-matrix / Kt feed.
// ---------------------------------------------------------------------------
__global__ __launch_bounds__(256) void k_proj32(const float* __restrict__ x,
                                                const float* __restrict__ w,
                                                const float* __restrict__ bias,
                                                u16* __restrict__ outb) {
  __shared__ float lw[CQK * CC];  // 32 KB
  const int tid = threadIdx.x;
  for (int e = tid; e < CQK * CC; e += 256) lw[e] = w[e];
  __syncthreads();

  const int g = blockIdx.x * 256 + tid;   // 0 .. B*N-1
  const int b = g >> 12;                  // /N
  const int n = g & (NN - 1);

  float acc[CQK];
#pragma unroll
  for (int o = 0; o < CQK; ++o) acc[o] = bias[o];

  for (int c = 0; c < CC; ++c) {
    const float xv = x[((size_t)b * CC + c) * NN + n];  // coalesced over tid
#pragma unroll
    for (int o = 0; o < CQK; ++o) acc[o] = fmaf(lw[o * CC + c], xv, acc[o]);
  }

  u16* dst = outb + (size_t)g * CQK;
#pragma unroll
  for (int o = 0; o < CQK; ++o) dst[o] = f2bf(acc[o]);
}

// ---------------------------------------------------------------------------
// K2: V projection. v[b][c][n] = sum_c' wv[c][c']*x[b][c'][n] + bv[c], bf16.
// ---------------------------------------------------------------------------
__global__ __launch_bounds__(256) void k_projV(const float* __restrict__ x,
                                               const float* __restrict__ wv,
                                               const float* __restrict__ bv,
                                               u16* __restrict__ vb) {
  __shared__ float lw[16 * CC];  // 16 KB
  const int tid = threadIdx.x;
  const int c0 = blockIdx.y * 16;
  const int b  = blockIdx.z;
  const int n  = blockIdx.x * 256 + tid;

  for (int e = tid; e < 16 * CC; e += 256)
    lw[e] = wv[(size_t)(c0 + (e >> 8)) * CC + (e & (CC - 1))];
  __syncthreads();

  float acc[16];
#pragma unroll
  for (int cc = 0; cc < 16; ++cc) acc[cc] = bv[c0 + cc];

  for (int cp = 0; cp < CC; ++cp) {
    const float xv = x[((size_t)b * CC + cp) * NN + n];
#pragma unroll
    for (int cc = 0; cc < 16; ++cc) acc[cc] = fmaf(lw[cc * CC + cp], xv, acc[cc]);
  }
#pragma unroll
  for (int cc = 0; cc < 16; ++cc)
    vb[((size_t)b * CC + c0 + cc) * NN + n] = f2bf(acc[cc]);
}

// ---------------------------------------------------------------------------
// K3: energy = Q * Kt^T. One v_wmma_f32_16x16x32_bf16 per 16x16 tile
// (K=32 == CQK in a single instruction). 8 waves/block -> 32(i) x 64(j).
// ---------------------------------------------------------------------------
__global__ __launch_bounds__(256) void k_energy(const u16* __restrict__ qb,
                                                const u16* __restrict__ kb,
                                                float* __restrict__ att) {
  const int tid  = threadIdx.x;
  const int wave = tid >> 5;
  const int lane = tid & 31;
  const int half = lane >> 4;
  const int l16  = lane & 15;

  const int b  = blockIdx.z;
  const int i0 = blockIdx.x * 32 + (wave >> 2) * 16;
  const int j0 = blockIdx.y * 64 + (wave & 3) * 16;

  const v16bf a = load_a16(qb + ((size_t)b * NN + (i0 + l16)) * CQK, half);

  // B: Kt rows (stored N x 32, same layout as Q); lane n = column j0+n,
  // K = half*16 + t, contiguous in the Kt row.
  const u16* krow = kb + ((size_t)b * NN + (j0 + l16)) * CQK + half * 16;
  const v16bf bm = __builtin_bit_cast(v16bf, *(const v16u*)krow);

  v8f c = {};
  c = __builtin_amdgcn_wmma_f32_16x16x32_bf16(false, a, false, bm,
                                              (short)0, c, false, false);

  // D: element v -> row i0 + v + 8*half, col j0 + l16
#pragma unroll
  for (int v = 0; v < 8; ++v) {
    const int i = i0 + v + half * 8;
    att[((size_t)b * NN + i) * NN + j0 + l16] = c[v];
  }
}

// ---------------------------------------------------------------------------
// K4: row softmax in place over att rows of length N=4096. 1 block per row.
// ---------------------------------------------------------------------------
__global__ __launch_bounds__(256) void k_softmax(float* __restrict__ att) {
  __shared__ float red[256];
  const int tid = threadIdx.x;
  float* p = att + (size_t)blockIdx.x * NN;

  float vals[16];
  float vmax = -3.402823466e+38f;
#pragma unroll
  for (int t = 0; t < 16; ++t) {
    vals[t] = p[tid + t * 256];
    vmax = fmaxf(vmax, vals[t]);
  }
  red[tid] = vmax;
  __syncthreads();
  for (int s = 128; s > 0; s >>= 1) {
    if (tid < s) red[tid] = fmaxf(red[tid], red[tid + s]);
    __syncthreads();
  }
  const float m = red[0];
  __syncthreads();

  float sum = 0.f;
#pragma unroll
  for (int t = 0; t < 16; ++t) {
    vals[t] = __expf(vals[t] - m);
    sum += vals[t];
  }
  red[tid] = sum;
  __syncthreads();
  for (int s = 128; s > 0; s >>= 1) {
    if (tid < s) red[tid] += red[tid + s];
    __syncthreads();
  }
  const float inv = 1.0f / red[0];
#pragma unroll
  for (int t = 0; t < 16; ++t) p[tid + t * 256] = vals[t] * inv;
}

// ---------------------------------------------------------------------------
// K5 helper: one j-step of out = V * P^T for two c-tiles of one wave.
// tile: 16x32 f32 P tile in LDS. Each lane reads 16 consecutive f32
// (4x ds_load_b128), converts to bf16 -> WMMA B operand.
// ---------------------------------------------------------------------------
__device__ __forceinline__ void av_step(const float* __restrict__ tile,
                                        const u16* __restrict__ va,
                                        const u16* __restrict__ vbr,
                                        int koff, int half, int l16,
                                        v8f& acc0, v8f& acc1) {
  const v16f pf = *(const v16f*)(tile + l16 * 32 + half * 16);
  v16u bu;
#pragma unroll
  for (int t = 0; t < 16; ++t) bu[t] = f2bf(pf[t]);
  const v16bf bm = __builtin_bit_cast(v16bf, bu);

  const v16bf a0 = load_a16(va + koff, half);
  acc0 = __builtin_amdgcn_wmma_f32_16x16x32_bf16(false, a0, false, bm,
                                                 (short)0, acc0, false, false);
  const v16bf a1 = load_a16(vbr + koff, half);
  acc1 = __builtin_amdgcn_wmma_f32_16x16x32_bf16(false, a1, false, bm,
                                                 (short)0, acc1, false, false);
}

// ---------------------------------------------------------------------------
// K5: out[b][c][i] = gamma * sum_j V[b][c][j]*P[b][i][j] + x[b][c][i].
// One block owns i-block of 16 and ALL 256 channels (8 waves x 2 c-tiles),
// so the P tile is read from HBM exactly once. The 16x32 f32 P tile is
// fetched by the Tensor Data Mover (double-buffered, TENSORcnt-tracked)
// when the toolchain exposes the TDM builtin.
// ---------------------------------------------------------------------------
__global__ __launch_bounds__(256) void k_av(const u16* __restrict__ vb,
                                            const float* __restrict__ att,
                                            const float* __restrict__ x,
                                            const float* __restrict__ gamma,
                                            float* __restrict__ out) {
  __shared__ __align__(128) float Pt[2][16 * 32];
  const int tid  = threadIdx.x;
  const int wave = tid >> 5;
  const int lane = tid & 31;
  const int half = lane >> 4;
  const int l16  = lane & 15;

  const int b  = blockIdx.y;
  const int i0 = blockIdx.x * 16;
  const int c0a = wave * 32;
  const int c0b = wave * 32 + 16;
  const float g = gamma[0];

  const u16* va  = vb + ((size_t)b * CC + (c0a + l16)) * NN;
  const u16* vbr = vb + ((size_t)b * CC + (c0b + l16)) * NN;

  v8f acc0 = {}, acc1 = {};

#if __has_builtin(__builtin_amdgcn_tensor_load_to_lds) && \
    __has_builtin(__builtin_amdgcn_s_wait_tensorcnt)
  // ---- TDM path: descriptor-driven 2D tile DMA (16 rows x 32 f32) ----
  const bool issuer = (wave == 0);
  // D# group1: wg_mask=0, data_size=4B(code 2), no pad/iterate,
  // tensor_dim0=N, tensor_dim1=N, tile_dim0=32, tile_dim1=16,
  // tensor_dim0_stride=N (elements).
  const v8i32 g1 = { 0x00020000, (NN << 16), (NN << 16), (32 << 16),
                     16, NN, 0, 0 };
  const v4i32 gz4 = { 0, 0, 0, 0 };
  const v8i32 gz8 = { 0, 0, 0, 0, 0, 0, 0, 0 };
  unsigned lds_base[2];
  lds_base[0] = (unsigned)(size_t)(&Pt[0][0]);
  lds_base[1] = (unsigned)(size_t)(&Pt[1][0]);
  // byte address of P[b][i0][0]
  const unsigned long long row_base =
      (unsigned long long)(size_t)(att + ((size_t)b * NN + i0) * NN);

  if (issuer) {
    const unsigned long long ga = row_base;  // tile k=0
    const v4u32 g0 = { 1u, lds_base[0], (unsigned)ga,
                       (unsigned)((ga >> 32) & 0x1FFFFFFu) | 0x80000000u };
    __builtin_amdgcn_tensor_load_to_lds(g0, g1, gz4, gz4, gz8, 0);
  }
  for (int k = 0; k < NN / 32; ++k) {
    const int cur = k & 1;
    if (issuer) {
      if (k + 1 < NN / 32) {
        const unsigned long long ga = row_base + (unsigned long long)(k + 1) * 128;
        const v4u32 g0 = { 1u, lds_base[cur ^ 1], (unsigned)ga,
                           (unsigned)((ga >> 32) & 0x1FFFFFFu) | 0x80000000u };
        __builtin_amdgcn_tensor_load_to_lds(g0, g1, gz4, gz4, gz8, 0);
        __builtin_amdgcn_s_wait_tensorcnt(1);  // tile k done (in-order)
      } else {
        __builtin_amdgcn_s_wait_tensorcnt(0);
      }
    }
    __syncthreads();                 // Pt[cur] visible to all waves
    av_step(&Pt[cur][0], va, vbr, k * 32, half, l16, acc0, acc1);
    __syncthreads();                 // all done with Pt[cur] before reuse
  }
#else
  // ---- Fallback: cooperative VGPR load of the P tile ----
  for (int k = 0; k < NN / 32; ++k) {
    const int j0 = k * 32;
    __syncthreads();
    for (int e = tid; e < 512; e += 256)
      Pt[0][e] = att[((size_t)b * NN + (i0 + (e >> 5))) * NN + j0 + (e & 31)];
    __syncthreads();
    av_step(&Pt[0][0], va, vbr, j0, half, l16, acc0, acc1);
  }
#endif

  // Epilogue: out = gamma*acc + x
#pragma unroll
  for (int v = 0; v < 8; ++v) {
    const int c = c0a + v + half * 8;
    const size_t idx = ((size_t)b * CC + c) * NN + i0 + l16;
    out[idx] = fmaf(g, acc0[v], x[idx]);
  }
#pragma unroll
  for (int v = 0; v < 8; ++v) {
    const int c = c0b + v + half * 8;
    const size_t idx = ((size_t)b * CC + c) * NN + i0 + l16;
    out[idx] = fmaf(g, acc1[v], x[idx]);
  }
}

// ---------------------------------------------------------------------------
extern "C" void kernel_launch(void* const* d_in, const int* in_sizes, int n_in,
                              void* d_out, int out_size, void* d_ws, size_t ws_size,
                              hipStream_t stream) {
  (void)in_sizes; (void)n_in; (void)out_size; (void)ws_size;

  const float* x     = (const float*)d_in[0];
  const float* wq    = (const float*)d_in[1];
  const float* bq    = (const float*)d_in[2];
  const float* wk    = (const float*)d_in[3];
  const float* bk    = (const float*)d_in[4];
  const float* wv    = (const float*)d_in[5];
  const float* bv    = (const float*)d_in[6];
  const float* gamma = (const float*)d_in[7];

  float* out = (float*)d_out;                      // B*C*N floats
  float* att = out + (size_t)BB * CC * NN;         // B*N*N floats

  u16* qb = (u16*)d_ws;                            // B*N*32 bf16
  u16* kb = qb + (size_t)BB * NN * CQK;            // B*N*32 bf16 (K transposed)
  u16* vbuf = kb + (size_t)BB * NN * CQK;          // B*C*N bf16

  // Projections
  k_proj32<<<(BB * NN) / 256, 256, 0, stream>>>(x, wq, bq, qb);
  k_proj32<<<(BB * NN) / 256, 256, 0, stream>>>(x, wk, bk, kb);
  k_projV<<<dim3(NN / 256, CC / 16, BB), 256, 0, stream>>>(x, wv, bv, vbuf);

  // energy -> softmax (in place in the attention output) -> AV + epilogue
  k_energy<<<dim3(NN / 32, NN / 64, BB), 256, 0, stream>>>(qb, kb, att);
  k_softmax<<<BB * NN, 256, 0, stream>>>(att);
  k_av<<<dim3(NN / 16, BB), 256, 0, stream>>>(vbuf, att, x, gamma, out);
}
